// BertGraphSelfAttention_34505767256981
// MI455X (gfx1250) — matrix operand order repaired and checked by reading, not compiled, not run on hardware
//
#include <hip/hip_runtime.h>
#include <hip/hip_bf16.h>

typedef _Float16 h16;
typedef __attribute__((ext_vector_type(16))) _Float16 v16h;
typedef __attribute__((ext_vector_type(8)))  float    v8f;
typedef __attribute__((ext_vector_type(4)))  float    f32x4;
typedef int v4i_vs __attribute__((__vector_size__(16)));   // matches builtin's V4i

union AF { v16h v; h16 h[16]; f32x4 q[2]; };
union CF { v8f  v; float f[8]; };

// ---- CDNA5 async global->LDS copy (ASYNCcnt path), with safe fallback ----

#if defined(__gfx1250__) && __has_builtin(__builtin_amdgcn_global_load_async_to_lds_b128)
#define ASYNC_LDS 1
#else
#define ASYNC_LDS 0
#endif

__device__ __forceinline__ void cp16_async(void* lds, const void* glb) {
#if ASYNC_LDS
  __builtin_amdgcn_global_load_async_to_lds_b128(
      (__attribute__((address_space(1))) v4i_vs*)glb,
      (__attribute__((address_space(3))) v4i_vs*)lds, 0, 0);
#else
  *(f32x4*)lds = *(const f32x4*)glb;
#endif
}

__device__ __forceinline__ void async_wait() {
#if ASYNC_LDS
#if __has_builtin(__builtin_amdgcn_s_wait_asynccnt)
  __builtin_amdgcn_s_wait_asynccnt(0);
#else
  asm volatile("s_wait_asynccnt 0" ::: "memory");
#endif
#endif
}

// ---- WMMA helpers (gfx1250, wave32) -------------------------------------

__device__ __forceinline__ v8f wmma_f16(v16h a, v16h b, v8f c) {
  // D = A(16x32,f16) * B(32x16,f16) + C(16x16,f32)
  return __builtin_amdgcn_wmma_f32_16x16x32_f16(
      /*neg_a=*/false, a, /*neg_b=*/false, b,
      /*c_mod=*/(short)0, c, /*reuse_a=*/false, /*reuse_b=*/false);
}

// A fragment 16x32 f16 from a row-major tile (ld in halfs).
// Lanes 0-15: M=lane, K blocks {0..7,16..23}; lanes 16-31: M=lane-16, K {8..15,24..31}.
__device__ __forceinline__ v16h load_a_frag(const h16* tile, int ld) {
  int lane = threadIdx.x & 31;
  int row  = lane & 15;
  int kb   = (lane >> 4) << 3;   // 0 or 8
  AF f;
  f.q[0] = *(const f32x4*)(tile + (size_t)row * ld + kb);
  f.q[1] = *(const f32x4*)(tile + (size_t)row * ld + 16 + kb);
  return f.v;
}

// B fragment 32x16 f16, sourced from an N-major (transposed) tile Bt[N][K].
// Lanes 0-15: N=lane, K blocks {0..7,16..23}; lanes 16-31: N=lane-16, K {8..15,24..31}.
__device__ __forceinline__ v16h load_bt_frag(const h16* tileT, int ld) {
  int lane = threadIdx.x & 31;
  int n    = lane & 15;
  int kb   = (lane >> 4) << 3;   // 0 or 8
  AF f;
  f.q[0] = *(const f32x4*)(tileT + (size_t)n * ld + kb);
  f.q[1] = *(const f32x4*)(tileT + (size_t)n * ld + kb + 16);
  return f.v;
}

// ---- Kernel 1: permute [b,m,seq,dim] -> [b*seq, m, dim], f32 -> f16 ------

__global__ __launch_bounds__(256) void prep_hs(const float* __restrict__ src,
                                               h16* __restrict__ dst) {
  size_t i = (size_t)blockIdx.x * 256 + threadIdx.x;   // exact grid
  int d = (int)(i & 511);
  size_t r = i >> 9;
  int j = (int)(r % 36); r /= 36;          // node index
  int s = (int)(r & 127);                  // seq index
  int bb = (int)(r >> 7);                  // batch
  dst[i] = (h16)src[((((size_t)bb * 36 + j) * 128 + s) << 9) + d];
}

// ---- Kernel 2: double-buffered WMMA GEMM  C = A(f16) @ W(f32) + bias -----

#define BM 128
#define BN 128
#define BKK 32

__global__ __launch_bounds__(256) void gemm_bias(const h16* __restrict__ A,
                                                 const float* __restrict__ W,
                                                 const float* __restrict__ bias,
                                                 h16* __restrict__ C,
                                                 int M, int N, int K) {
  __shared__ __align__(16) h16 As[2][BM][BKK];   // 16 KB double-buffered
  __shared__ __align__(16) h16 Wt[2][BN][BKK];   // 16 KB (N-major = B^T tiles)
  int bm = blockIdx.x, bn = blockIdx.y;
  int t = threadIdx.x;
  int w = t >> 5;
  int wm = (w >> 1) * 32;          // wave M offset (0,32,64,96)
  int wn = (w & 1) * 64;           // wave N offset (0,64)
  int lane = t & 31;
  int cr = (lane >> 4) * 8, cc = lane & 15;
  int ar = t >> 1, aseg = (t & 1) * 16;            // A staging assignment
  const h16* Arow = A + (size_t)(bm * BM + ar) * K + aseg;

  auto stage = [&](int buf, int k0) {
    // A tile: async global->LDS, 2x16B per thread
    cp16_async(&As[buf][ar][aseg],     Arow + k0);
    cp16_async(&As[buf][ar][aseg + 8], Arow + k0 + 8);
    // W^T tile: through VGPRs (needs f32->f16 convert)
    for (int i = t; i < BN * BKK; i += 256) {
      int n = i & (BN - 1);
      int k = i >> 7;
      Wt[buf][n][k] = (h16)W[(size_t)(k0 + k) * N + bn * BN + n];
    }
  };

  v8f zz = {};
  CF acc[2][4];
  for (int i = 0; i < 2; i++)
    for (int j = 0; j < 4; j++) acc[i][j].v = zz;

  stage(0, 0);
  async_wait();
  __syncthreads();

  int nk = K / BKK;
  for (int ki = 0; ki < nk; ki++) {
    int cur = ki & 1;
    if (ki + 1 < nk) stage(cur ^ 1, (ki + 1) * BKK);   // prefetch next tile

    v16h af0 = load_a_frag(&As[cur][wm][0], BKK);
    v16h af1 = load_a_frag(&As[cur][wm + 16][0], BKK);
    for (int j = 0; j < 4; j++) {
      v16h bf = load_bt_frag(&Wt[cur][wn + j * 16][0], BKK);
      acc[0][j].v = wmma_f16(af0, bf, acc[0][j].v);
      acc[1][j].v = wmma_f16(af1, bf, acc[1][j].v);
    }
    if (ki + 1 < nk) { async_wait(); __syncthreads(); }
  }

  // epilogue: add bias, convert, store
  for (int j = 0; j < 4; j++) {
    int col = bn * BN + wn + j * 16 + cc;
    float bv = bias[col];
    for (int i = 0; i < 2; i++)
      for (int v = 0; v < 8; v++) {
        int row = bm * BM + wm + i * 16 + cr + v;
        C[(size_t)row * N + col] = (h16)(acc[i][j].f[v] + bv);
      }
  }
}

// ---- Kernel 3: branch-1 graph attention, one wave per (n, h) -------------
// Q,K,V: [2048*36, 512] f16, head h occupies cols [h*128, h*128+128)
// out hs2: [b*m, seq, 512] f16 (transpose fused into store)

__global__ __launch_bounds__(32) void attn_graph(const h16* __restrict__ Q,
                                                 const h16* __restrict__ K,
                                                 const h16* __restrict__ V,
                                                 const float* __restrict__ simg,
                                                 const float* __restrict__ am,
                                                 h16* __restrict__ hs2) {
  int n = blockIdx.x;        // 0..2047  (= bb*128 + s)
  int h = blockIdx.y;        // 0..3
  int lane = threadIdx.x;
  int bb = n >> 7, s = n & 127;
  int cr = (lane >> 4) * 8, cc = lane & 15;

  __shared__ __align__(16) h16   Qs[48][128];   // 12 KB (rows 36..47 zero)
  __shared__ __align__(16) h16   Ks[48][128];   // 12 KB
  __shared__ __align__(16) h16   Vt[128][64];   // 16 KB  Vt[d][j] (j 36..63 zero)
  __shared__ __align__(16) float Ss[48][48];    //  9 KB  scores
  __shared__ __align__(16) h16   Ps[48][64];    //  6 KB  probs (K-padded)

  const h16* Qg = Q + ((size_t)n * 36) * 512 + (size_t)h * 128;
  const h16* Kg = K + ((size_t)n * 36) * 512 + (size_t)h * 128;
  const h16* Vg = V + ((size_t)n * 36) * 512 + (size_t)h * 128;

  // rows 0..35: async copy; rows 36..47: zero fill (uniform trip counts)
  for (int idx = lane; idx < 36 * 16; idx += 32) {
    int r = idx >> 4, seg = (idx & 15) * 8;
    cp16_async(&Qs[r][seg], Qg + (size_t)r * 512 + seg);
    cp16_async(&Ks[r][seg], Kg + (size_t)r * 512 + seg);
  }
  f32x4 z4 = {0.f, 0.f, 0.f, 0.f};
  for (int idx = lane; idx < 12 * 16; idx += 32) {
    int r = 36 + (idx >> 4), seg = (idx & 15) * 8;
    *(f32x4*)&Qs[r][seg] = z4;
    *(f32x4*)&Ks[r][seg] = z4;
  }
  for (int idx = lane; idx < 128 * 64; idx += 32) {
    int d = idx >> 6, j = idx & 63;
    Vt[d][j] = (j < 36) ? Vg[(size_t)j * 512 + d] : (h16)0.f;
  }
  for (int idx = lane; idx < 48 * 64; idx += 32) ((h16*)Ps)[idx] = (h16)0.f;
  async_wait();
  __syncthreads();

  // scores = Q @ K^T   (48x48, padded)
  v8f zz = {};
  CF accs[3][3];
  for (int mf = 0; mf < 3; mf++)
    for (int nf = 0; nf < 3; nf++) accs[mf][nf].v = zz;
  for (int kk = 0; kk < 4; kk++) {
    v16h qa[3];
    for (int mf = 0; mf < 3; mf++) qa[mf] = load_a_frag(&Qs[mf * 16][kk * 32], 128);
    for (int nf = 0; nf < 3; nf++) {
      v16h bf = load_bt_frag(&Ks[nf * 16][kk * 32], 128);  // Bt of K^T is K itself
      for (int mf = 0; mf < 3; mf++) accs[mf][nf].v = wmma_f16(qa[mf], bf, accs[mf][nf].v);
    }
  }
  const float scale = 0.08838834764831845f;   // 1/sqrt(128)
  for (int mf = 0; mf < 3; mf++)
    for (int nf = 0; nf < 3; nf++)
      for (int v = 0; v < 8; v++) {
        int qi = mf * 16 + cr + v, ki = nf * 16 + cc;
        float val;
        if (qi < 36 && ki < 36) {
          float mk = am[((size_t)bb * 36 + ki) * 128 + s];
          float sg = simg[(((size_t)n * 4 + h) * 36 + qi) * 36 + ki];
          if (mk == 0.f) sg = 0.f;
          val = accs[mf][nf].f[v] * scale + (1.f - sg) * -10000.f;
        } else {
          val = -1e30f;
        }
        Ss[qi][ki] = val;
      }
  __syncthreads();

  // masked softmax over 36 keys
  for (int r = lane; r < 36; r += 32) {
    float mx = -1e30f;
    for (int k2 = 0; k2 < 36; k2++) mx = fmaxf(mx, Ss[r][k2]);
    float sum = 0.f;
    for (int k2 = 0; k2 < 36; k2++) sum += __expf(Ss[r][k2] - mx);
    float inv = 1.f / sum;
    for (int k2 = 0; k2 < 36; k2++) Ps[r][k2] = (h16)(__expf(Ss[r][k2] - mx) * inv);
  }
  __syncthreads();

  // ctx = P @ V, store transposed into hs2[(bb*36+q)*128 + s][h*128 + d]
  v16h pa[3][2];
  for (int mf = 0; mf < 3; mf++)
    for (int kk = 0; kk < 2; kk++) pa[mf][kk] = load_a_frag(&Ps[mf * 16][kk * 32], 64);
  for (int nf = 0; nf < 8; nf++) {
    CF a3[3];
    for (int mf = 0; mf < 3; mf++) a3[mf].v = zz;
    for (int kk = 0; kk < 2; kk++) {
      v16h bf = load_bt_frag(&Vt[nf * 16][kk * 32], 64);
      for (int mf = 0; mf < 3; mf++) a3[mf].v = wmma_f16(pa[mf][kk], bf, a3[mf].v);
    }
    for (int mf = 0; mf < 3; mf++)
      for (int v = 0; v < 8; v++) {
        int qi = mf * 16 + cr + v;
        if (qi < 36) {
          int d = nf * 16 + cc;
          hs2[(((size_t)bb * 36 + qi) * 128 + s) * 512 + (size_t)h * 128 + d] =
              (h16)a3[mf].f[v];
        }
      }
  }
}

// ---- Kernel 4: branch-2 temporal attention with Shaw relative positions --
// grid (576, 4, 4): (n2, head, 32-row block). 64 threads = 2 waves.

__global__ __launch_bounds__(64) void attn_seq(const h16* __restrict__ Q,
                                               const h16* __restrict__ K,
                                               const h16* __restrict__ V,
                                               const float* __restrict__ am,
                                               const float* __restrict__ relk,
                                               const float* __restrict__ relv,
                                               float* __restrict__ out) {
  int n2 = blockIdx.x;       // 0..575 (= bb*36 + mj)
  int h  = blockIdx.y;       // 0..3
  int mb = blockIdx.z;       // 0..3  rows mb*32
  int t = threadIdx.x;
  int lane = t & 31, w = t >> 5;
  int cr = (lane >> 4) * 8, cc = lane & 15;

  __shared__ __align__(16) h16 Qs[32][128];            // 8 KB; reused for probs
  __shared__ __align__(16) h16 KV[128][128];           // 32 KB; K^T-source, then V^T
  __shared__ __align__(16) unsigned char SR[32 * 128 * 4]; // 16 KB; scores, then rel_v^T
  __shared__ __align__(16) h16 Pb[32][64];             // 4 KB; bucketed probs (K pad 33->64)
  float (*Ss)[128] = (float (*)[128])SR;
  h16   (*Rvt)[64] = (h16 (*)[64])SR;

  const h16* Qg = Q + ((size_t)n2 * 128 + mb * 32) * 512 + (size_t)h * 128;
  const h16* Kg = K + (size_t)n2 * 128 * 512 + (size_t)h * 128;
  const h16* Vg = V + (size_t)n2 * 128 * 512 + (size_t)h * 128;

  for (int idx = t; idx < 32 * 16; idx += 64) {
    int r = idx >> 4, seg = (idx & 15) * 8;
    cp16_async(&Qs[r][seg], Qg + (size_t)r * 512 + seg);
  }
  for (int idx = t; idx < 128 * 16; idx += 64) {
    int r = idx >> 4, seg = (idx & 15) * 8;
    cp16_async(&KV[r][seg], Kg + (size_t)r * 512 + seg);
  }
  async_wait();
  __syncthreads();

  // raw scores = Q2 @ K2^T  (each wave: 16 rows x 128 cols)
  v8f zz = {};
  {
    v16h qa[4];
    for (int kk = 0; kk < 4; kk++) qa[kk] = load_a_frag(&Qs[w * 16][kk * 32], 128);
    for (int nf = 0; nf < 8; nf++) {
      CF acc; acc.v = zz;
      for (int kk = 0; kk < 4; kk++)
        acc.v = wmma_f16(qa[kk], load_bt_frag(&KV[nf * 16][kk * 32], 128), acc.v);
      for (int v = 0; v < 8; v++) Ss[w * 16 + cr + v][nf * 16 + cc] = acc.f[v];
    }
  }
  __syncthreads();

  // add q2 . rel_k(clip(k-q)) term, scale, mask  (VALU: only 33 distances)
  const float scale = 0.08838834764831845f;
  {
    int q = t >> 1;                 // 0..31
    int qg = mb * 32 + q;
    int k0 = (t & 1) * 64;
    for (int k2 = k0; k2 < k0 + 64; k2++) {
      int r = k2 - qg;
      r = (r < -16) ? -16 : (r > 16 ? 16 : r);
      const float* rk = relk + (size_t)(r + 16) * 128;
      float dot = 0.f;
      for (int d = 0; d < 128; d++) dot += (float)Qs[q][d] * rk[d];
      float val = (Ss[q][k2] + dot) * scale;
      val += (1.f - am[(size_t)n2 * 128 + k2]) * -10000.f;
      Ss[q][k2] = val;
    }
  }
  __syncthreads();

  // softmax + distance-bucket compaction of probs (rel_v term -> GEMM)
  if (t < 32) {
    int q = t, qg = mb * 32 + q;
    float mx = -1e30f;
    for (int k2 = 0; k2 < 128; k2++) mx = fmaxf(mx, Ss[q][k2]);
    float sum = 0.f;
    for (int k2 = 0; k2 < 128; k2++) sum += __expf(Ss[q][k2] - mx);
    float inv = 1.f / sum;
    for (int r = 0; r < 64; r++) Pb[q][r] = (h16)0.f;
    float b0 = 0.f, b32 = 0.f;
    for (int k2 = 0; k2 < 128; k2++) {
      float p = __expf(Ss[q][k2] - mx) * inv;
      Qs[q][k2] = (h16)p;                       // probs overwrite Q
      int r = k2 - qg;
      if (r <= -16) b0 += p;
      else if (r >= 16) b32 += p;
      else Pb[q][r + 16] = (h16)p;              // unique k per interior bucket
    }
    Pb[q][0] = (h16)b0;
    Pb[q][32] = (h16)b32;
  }
  __syncthreads();

  // stage V^T into KV and rel_v^T (padded to K=64) into SR
  for (int idx = t; idx < 128 * 128; idx += 64) {
    int d = idx >> 7, k2 = idx & 127;
    KV[d][k2] = Vg[(size_t)k2 * 512 + d];
  }
  for (int idx = t; idx < 128 * 64; idx += 64) {
    int d = idx >> 6, r = idx & 63;
    Rvt[d][r] = (r < 33) ? (h16)relv[(size_t)r * 128 + d] : (h16)0.f;
  }
  __syncthreads();

  // ctx2 = P @ V  +  Pb @ rel_v   (fused into one accumulator chain)
  v16h pa[4], pr[2];
  for (int kk = 0; kk < 4; kk++) pa[kk] = load_a_frag(&Qs[w * 16][kk * 32], 128);
  for (int kk = 0; kk < 2; kk++) pr[kk] = load_a_frag(&Pb[w * 16][kk * 32], 64);
  for (int nf = 0; nf < 8; nf++) {
    CF acc; acc.v = zz;
    for (int kk = 0; kk < 4; kk++)
      acc.v = wmma_f16(pa[kk], load_bt_frag(&KV[nf * 16][kk * 32], 128), acc.v);
    for (int kk = 0; kk < 2; kk++)
      acc.v = wmma_f16(pr[kk], load_bt_frag(&Rvt[nf * 16][kk * 32], 64), acc.v);
    for (int v = 0; v < 8; v++) {
      int s = mb * 32 + w * 16 + cr + v;
      out[((size_t)n2 * 128 + s) * 512 + (size_t)h * 128 + nf * 16 + cc] = acc.f[v];
    }
  }
}

// ---- Host launch ---------------------------------------------------------

extern "C" void kernel_launch(void* const* d_in, const int* in_sizes, int n_in,
                              void* d_out, int out_size, void* d_ws, size_t ws_size,
                              hipStream_t stream) {
  (void)in_sizes; (void)n_in; (void)out_size; (void)ws_size;
  const float* hidden = (const float*)d_in[0];
  const float* am     = (const float*)d_in[1];
  const float* simg   = (const float*)d_in[2];
  const float* Wq1 = (const float*)d_in[3];  const float* bq1 = (const float*)d_in[4];
  const float* Wk1 = (const float*)d_in[5];  const float* bk1 = (const float*)d_in[6];
  const float* Wv1 = (const float*)d_in[7];  const float* bv1 = (const float*)d_in[8];
  const float* Wq2 = (const float*)d_in[9];  const float* bq2 = (const float*)d_in[10];
  const float* Wk2 = (const float*)d_in[11]; const float* bk2 = (const float*)d_in[12];
  const float* Wv2 = (const float*)d_in[13]; const float* bv2 = (const float*)d_in[14];
  const float* relk = (const float*)d_in[15];
  const float* relv = (const float*)d_in[16];
  float* out = (float*)d_out;

  const size_t BUFE = 73728ull * 512;   // elements per f16 buffer
  h16* hsT = (h16*)d_ws;                // buffer 0: hsT, later hs2
  h16* q1  = hsT + BUFE;                // buffer 1: q1, later q2
  h16* k1  = q1 + BUFE;                 // buffer 2: k1, later k2
  h16* v1  = k1 + BUFE;                 // buffer 3: v1, later v2
  h16* hs2 = hsT;
  h16* q2 = q1; h16* k2 = k1; h16* v2 = v1;

  const int M = 73728, N = 512, Kd = 512;
  size_t total = (size_t)16 * 36 * 128 * 512;
  prep_hs<<<(unsigned)(total / 256), 256, 0, stream>>>(hidden, hsT);

  dim3 gg(M / BM, N / BN);
  gemm_bias<<<gg, 256, 0, stream>>>(hsT, Wq1, bq1, q1, M, N, Kd);
  gemm_bias<<<gg, 256, 0, stream>>>(hsT, Wk1, bk1, k1, M, N, Kd);
  gemm_bias<<<gg, 256, 0, stream>>>(hsT, Wv1, bv1, v1, M, N, Kd);

  attn_graph<<<dim3(2048, 4), 32, 0, stream>>>(q1, k1, v1, simg, am, hs2);

  gemm_bias<<<gg, 256, 0, stream>>>(hs2, Wq2, bq2, q2, M, N, Kd);
  gemm_bias<<<gg, 256, 0, stream>>>(hs2, Wk2, bk2, k2, M, N, Kd);
  gemm_bias<<<gg, 256, 0, stream>>>(hs2, Wv2, bv2, v2, M, N, Kd);

  attn_seq<<<dim3(576, 4, 4), 64, 0, stream>>>(q2, k2, v2, am, relk, relv, out);
}
